// Model_1133871366654
// MI455X (gfx1250) — matrix operand order repaired and checked by reading, not compiled, not run on hardware
//
#include <hip/hip_runtime.h>

// ---------------------------------------------------------------------------
// BiLSTM-CRF for gfx1250 (MI455X)
//   - WMMA f16 GEMMs (input projection + recurrence)
//   - persistent LSTM: W_hh B-fragments resident in VGPRs (128 regs/lane),
//     xg slabs (transposed [gate][row]) double-buffered into LDS via
//     global_load_async_to_lds_b128 / s_wait_asynccnt
//   - register-resident LSTM cell state, fast exp-based sigmoid/tanh
//   - wave-per-row Viterbi with LDS-resident emissions/mask/backpointers
// ---------------------------------------------------------------------------

typedef __attribute__((ext_vector_type(16))) _Float16 v16h;
typedef __attribute__((ext_vector_type(8)))  _Float16 v8h;
typedef __attribute__((ext_vector_type(8)))  float    v8f;

#define E_   128
#define H2_  128
#define G_   512            // 4*H2
#define T_   9
#define B_   128
#define S_   512
#define NT_  (B_ * S_)      // 65536 tokens

// --------------------------- CDNA5 async helpers ---------------------------
__device__ __forceinline__ unsigned lds_off32(const void* p) {
    // LDS aperture: generic addr[31:0] is the wave-relative LDS byte address
    return (unsigned)(unsigned long long)p;
}
__device__ __forceinline__ void async_copy_b128(void* lds_dst, const void* gsrc) {
    unsigned d = lds_off32(lds_dst);
    unsigned long long a = (unsigned long long)gsrc;
    asm volatile("global_load_async_to_lds_b128 %0, %1, off"
                 :: "v"(d), "v"(a) : "memory");
}
__device__ __forceinline__ void wait_async0() {
    asm volatile("s_wait_asynccnt 0x0" ::: "memory");
}

__device__ __forceinline__ float sigmoidf_(float v) { return 1.f / (1.f + __expf(-v)); }
__device__ __forceinline__ float tanhf_(float v) {
    float e = __expf(2.f * v);
    return 1.f - 2.f / (e + 1.f);
}

// ---------------------------------------------------------------------------
// K0: convert weights to f16 (row-major [2][512][128] for each of W_ih, W_hh)
// ---------------------------------------------------------------------------
__global__ void pack_weights(const float* __restrict__ Wih_f, const float* __restrict__ Wih_b,
                             const float* __restrict__ Whh_f, const float* __restrict__ Whh_b,
                             _Float16* __restrict__ Wih16, _Float16* __restrict__ Whh16) {
    int i = blockIdx.x * blockDim.x + threadIdx.x;   // 0 .. 2*G_*E_-1
    const int n = G_ * E_;
    if (i < n) {
        Wih16[i] = (_Float16)Wih_f[i];
        Whh16[i] = (_Float16)Whh_f[i];
    } else {
        Wih16[i] = (_Float16)Wih_b[i - n];
        Whh16[i] = (_Float16)Whh_b[i - n];
    }
}

// ---------------------------------------------------------------------------
// K1: input projection -> xg16 in LSTM-slab TRANSPOSED layout:
//     xg16[dir][bchunk][t][gate(512)][row(16)]  (f16)
//     One block = 16 consecutive tokens (same batch row b, 16 s values)
//     x 128 gates; 8 waves, one 16x16 WMMA output tile per wave.
// ---------------------------------------------------------------------------
__global__ __launch_bounds__(256)
void input_proj(const int* __restrict__ x, const float* __restrict__ embed,
                const _Float16* __restrict__ Wih16,
                const float* __restrict__ b_f, const float* __restrict__ b_b,
                _Float16* __restrict__ xg16) {
    __shared__ __align__(32) _Float16 sA[16 * 128];    // 16 tokens x 128 dims
    __shared__ __align__(32) _Float16 sW[128 * 128];   // 128 gates x 128 K

    const int tid   = threadIdx.x;
    const int tbase = blockIdx.x * 16;
    const int gb    = blockIdx.y;          // 0..7
    const int dir   = gb >> 2;
    const int g0    = (gb & 3) * 128;      // gate base within direction

    const int bidx  = tbase >> 9;          // batch row (constant for the tile)
    const int sbase = tbase & (S_ - 1);    // sequence base
    const int bc    = bidx >> 4;           // batch chunk 0..7
    const int row   = bidx & 15;           // row within chunk

    // Stage A: gather 16 embedding rows, convert to f16
    for (int i = 0; i < 8; ++i) {
        int e   = tid + i * 256;
        int r   = e >> 7, col = e & 127;
        int vi  = x[tbase + r];
        sA[e] = (_Float16)embed[(size_t)vi * E_ + col];
    }
    // Stage W tile: 128x128 f16, contiguous 32KB copy
    {
        const uint4* src = (const uint4*)(Wih16 + ((size_t)dir * G_ + g0) * E_);
        uint4*       dst = (uint4*)sW;
        for (int i = 0; i < 8; ++i) dst[tid + i * 256] = src[tid + i * 256];
    }
    __syncthreads();

    const int lane = tid & 31;
    const int w    = tid >> 5;
    const int kb   = lane >> 4;
    const int ln   = lane & 15;

    v8f acc = {};
#pragma unroll
    for (int k0 = 0; k0 < 128; k0 += 32) {
        v8h lo = *(const v8h*)&sA[ln * 128 + k0 + kb * 16];
        v8h hi = *(const v8h*)&sA[ln * 128 + k0 + 8 + kb * 16];
        v16h a;
        for (int e = 0; e < 8; ++e) { a[e] = lo[e]; a[8 + e] = hi[e]; }
        v16h b = *(const v16h*)&sW[(w * 16 + ln) * 128 + k0 + kb * 16];
        acc = __builtin_amdgcn_wmma_f32_16x16x32_f16(false, a, false, b,
                                                     (short)0, acc, false, false);
    }

    const float* bias = dir ? b_b : b_f;
    const float  bv   = bias[g0 + w * 16 + ln];
    const int    g    = g0 + w * 16 + ln;
    // slab layout: [t][gate][row]; s varies with accumulator element
    const size_t blkbase = (size_t)(dir * 8 + bc) * S_ * G_ * 16;
#pragma unroll
    for (int r = 0; r < 8; ++r) {
        int s = sbase + r + kb * 8;
        xg16[blkbase + ((size_t)s * G_ + g) * 16 + row] = (_Float16)(acc[r] + bv);
    }
}

// ---------------------------------------------------------------------------
// K2: persistent bidirectional LSTM. 16 blocks = 2 dirs x 8 batch chunks.
//     B-fragments of W_hh live in VGPRs (16 x v16h per lane) for all 512
//     steps; xg slabs double-buffered in LDS via async loads.
//     Wave w owns gate columns j = w*16+ln across all four gate groups
//     (nt = q*8+w): i/f/g/o accumulators + cell state stay in registers.
// ---------------------------------------------------------------------------
__global__ __launch_bounds__(256)
void lstm_kernel(const _Float16* __restrict__ xg16, const _Float16* __restrict__ Whh16,
                 _Float16* __restrict__ hbuf16) {
    __shared__ __align__(32) _Float16 sh[16 * 128];        // h_{t-1} (f16)
    __shared__ __align__(32) _Float16 slab[2][G_ * 16];    // 2 x 16KB xg slabs

    const int tid   = threadIdx.x;
    const int dir   = blockIdx.x >> 3;
    const int bc    = blockIdx.x & 7;
    const int bbase = bc * 16;

    const int lane = tid & 31;
    const int w    = tid >> 5;
    const int kb   = lane >> 4;
    const int ln   = lane & 15;
    const int j    = w * 16 + ln;                          // hidden column owned

    // h0 = 0
    for (int i = 0; i < 8; ++i) sh[tid + i * 256] = (_Float16)0.f;

    // Loop-invariant B fragments: W_hh rows for gate tiles nt = q*8+w
    const _Float16* Wd = Whh16 + (size_t)dir * G_ * H2_;
    v16h bf[16];
#pragma unroll
    for (int q = 0; q < 4; ++q) {
        const int nt = q * 8 + w;
#pragma unroll
        for (int kk = 0; kk < 4; ++kk) {
            bf[q * 4 + kk] =
                *(const v16h*)&Wd[(size_t)(nt * 16 + ln) * H2_ + kk * 32 + kb * 16];
        }
    }

    const _Float16* xslab = xg16 + (size_t)(dir * 8 + bc) * S_ * G_ * 16;
    _Float16*       hot   = hbuf16 + (size_t)dir * NT_ * H2_;

    // Preload slab for t0
    {
        const int t0 = dir ? (S_ - 1) : 0;
        for (int i = 0; i < 4; ++i) {
            int e = tid + i * 256;                         // 16B chunk index
            async_copy_b128(&slab[0][e * 8], xslab + (size_t)t0 * (G_ * 16) + e * 8);
        }
    }
    wait_async0();
    __syncthreads();

    float creg[8];
#pragma unroll
    for (int r = 0; r < 8; ++r) creg[r] = 0.f;

    for (int step = 0; step < S_; ++step) {
        const int t = dir ? (S_ - 1 - step) : step;
        const _Float16* cur = slab[step & 1];
        _Float16*       nxt = slab[(step + 1) & 1];

        // Prefetch next timestep's xg slab (async, overlapped with compute)
        if (step + 1 < S_) {
            const int tn = dir ? (S_ - 2 - step) : (step + 1);
            for (int i = 0; i < 4; ++i) {
                int e = tid + i * 256;
                async_copy_b128(&nxt[e * 8], xslab + (size_t)tn * (G_ * 16) + e * 8);
            }
        }

        // A fragments from h_{t-1}
        v16h af[4];
#pragma unroll
        for (int kk = 0; kk < 4; ++kk) {
            int k0 = kk * 32;
            v8h lo = *(const v8h*)&sh[ln * 128 + k0 + kb * 16];
            v8h hi = *(const v8h*)&sh[ln * 128 + k0 + 8 + kb * 16];
            for (int e = 0; e < 8; ++e) { af[kk][e] = lo[e]; af[kk][8 + e] = hi[e]; }
        }

        // Four gate-group accumulators: q=0:i, 1:f, 2:g, 3:o, column j
        v8f acc[4];
#pragma unroll
        for (int q = 0; q < 4; ++q) {
            const int nt = q * 8 + w;
            // xg init: one contiguous v8h per accumulator (rows kb*8 .. kb*8+7)
            v8h xv = *(const v8h*)&cur[(nt * 16 + ln) * 16 + kb * 8];
#pragma unroll
            for (int r = 0; r < 8; ++r) acc[q][r] = (float)xv[r];
#pragma unroll
            for (int kk = 0; kk < 4; ++kk) {
                acc[q] = __builtin_amdgcn_wmma_f32_16x16x32_f16(false, af[kk], false,
                                                                bf[q * 4 + kk],
                                                                (short)0, acc[q],
                                                                false, false);
            }
        }

        // Register-resident LSTM cell update
        _Float16* hrow = hot + (size_t)(bbase + kb * 8) * S_ * H2_ + (size_t)t * H2_ + j;
#pragma unroll
        for (int r = 0; r < 8; ++r) {
            int   mm = r + kb * 8;
            float c  = creg[r];
            c = sigmoidf_(acc[1][r]) * c + sigmoidf_(acc[0][r]) * tanhf_(acc[2][r]);
            float h = sigmoidf_(acc[3][r]) * tanhf_(c);
            creg[r] = c;
            sh[mm * 128 + j] = (_Float16)h;
            hrow[(size_t)r * S_ * H2_] = (_Float16)h;
        }

        wait_async0();      // next slab fully landed
        __syncthreads();    // everyone's loads + sh writes visible
    }
}

// ---------------------------------------------------------------------------
// K3: emissions[tok][tag] = concat(h_f,h_b) . W_out[tag] + b_out[tag]  (T=9)
// ---------------------------------------------------------------------------
__global__ void emissions_kernel(const _Float16* __restrict__ hbuf16,
                                 const float* __restrict__ W_out,
                                 const float* __restrict__ b_out,
                                 float* __restrict__ em) {
    int idx = blockIdx.x * blockDim.x + threadIdx.x;
    if (idx >= NT_ * T_) return;
    int tok = idx / T_, tag = idx % T_;
    const _Float16* hf = hbuf16 + (size_t)tok * H2_;
    const _Float16* hb = hbuf16 + (size_t)NT_ * H2_ + (size_t)tok * H2_;
    const float*    wr = W_out + tag * 256;
    float s = b_out[tag];
    for (int k = 0; k < 128; k += 8) {
        v8h a = *(const v8h*)&hf[k];
        for (int e = 0; e < 8; ++e) s += (float)a[e] * wr[k + e];
    }
    for (int k = 0; k < 128; k += 8) {
        v8h a = *(const v8h*)&hb[k];
        for (int e = 0; e < 8; ++e) s += (float)a[e] * wr[128 + k + e];
    }
    em[idx] = s;
}

// ---------------------------------------------------------------------------
// K4: Viterbi decode. One wave (32 lanes) per batch row, 128 blocks.
//     Lane j < 9 owns tag column j; scores exchanged via __shfl.
//     Emissions, mask, and backpointer history all LDS-resident, so the
//     serial forward scan and the backtrack pointer-chase never touch
//     global memory latency.
//     d_out layout: paths as float [B*S] then best_scores [B].
// ---------------------------------------------------------------------------
__global__ __launch_bounds__(32)
void viterbi_kernel(const float* __restrict__ em, const int* __restrict__ mask,
                    const float* __restrict__ start, const float* __restrict__ trans,
                    const float* __restrict__ endt,
                    float* __restrict__ out) {
    __shared__ float sem[S_ * T_];          // 18.4KB emissions for this row
    __shared__ int   smask[S_];             //  2.0KB mask for this row
    __shared__ int   hist[(S_ - 1) * 16];   // 32.7KB backpointers (stride 16)

    const int b    = blockIdx.x;
    const int lane = threadIdx.x;
    const int j    = (lane < T_) ? lane : (T_ - 1);   // clamped tag column

    // Stage emissions + mask into LDS (coalesced)
    for (int k = lane; k < S_ * T_; k += 32) sem[k] = em[(size_t)b * S_ * T_ + k];
    for (int s = lane; s < S_; s += 32) smask[s] = mask[b * S_ + s];

    // Per-lane transition column trc[i] = trans[i][j]
    float trc[T_];
#pragma unroll
    for (int i = 0; i < T_; ++i) trc[i] = trans[i * T_ + j];

    float score = start[j] + sem[j];

    for (int s = 1; s < S_; ++s) {
        float emj = sem[s * T_ + j];
        float bv  = __shfl(score, 0, 32) + trc[0];
        int   bi  = 0;
#pragma unroll
        for (int i = 1; i < T_; ++i) {
            float v = __shfl(score, i, 32) + trc[i];
            if (v > bv) { bv = v; bi = i; }
        }
        if (lane < T_) hist[(s - 1) * 16 + lane] = bi;
        if (smask[s])  score = bv + emj;
    }

    // Argmax of final scores (broadcast via shuffles; all lanes agree)
    float fin = score + endt[j];
    float bs  = __shfl(fin, 0, 32);
    int   cur = 0;
#pragma unroll
    for (int i = 1; i < T_; ++i) {
        float v = __shfl(fin, i, 32);
        if (v > bs) { bs = v; cur = i; }
    }

    if (lane == 0) {
        out[(size_t)B_ * S_ + b] = bs;
        float* paths = out + (size_t)b * S_;
        int c = cur;
        for (int s = S_ - 1; s >= 1; --s) {
            int m = smask[s];
            paths[s] = m ? (float)c : -1.f;
            int prev = hist[(s - 1) * 16 + c];
            if (m) c = prev;
        }
        paths[0] = (float)c;
    }
}

// ---------------------------------------------------------------------------
extern "C" void kernel_launch(void* const* d_in, const int* in_sizes, int n_in,
                              void* d_out, int out_size, void* d_ws, size_t ws_size,
                              hipStream_t stream) {
    const int*   x     = (const int*)  d_in[0];
    const int*   mask  = (const int*)  d_in[1];
    const float* embed = (const float*)d_in[2];
    const float* Wih_f = (const float*)d_in[3];
    const float* Whh_f = (const float*)d_in[4];
    const float* b_f   = (const float*)d_in[5];
    const float* Wih_b = (const float*)d_in[6];
    const float* Whh_b = (const float*)d_in[7];
    const float* b_b   = (const float*)d_in[8];
    const float* W_out = (const float*)d_in[9];
    const float* b_out = (const float*)d_in[10];
    const float* start = (const float*)d_in[11];
    const float* trans = (const float*)d_in[12];
    const float* endt  = (const float*)d_in[13];

    char* p = (char*)d_ws;
    auto carve = [&p](size_t bytes) {
        char* r = p;
        p += (bytes + 255) & ~(size_t)255;
        return (void*)r;
    };
    _Float16* Wih16  = (_Float16*)carve((size_t)2 * G_ * E_  * sizeof(_Float16));
    _Float16* Whh16  = (_Float16*)carve((size_t)2 * G_ * H2_ * sizeof(_Float16));
    _Float16* xg16   = (_Float16*)carve((size_t)2 * NT_ * G_  * sizeof(_Float16));
    _Float16* hbuf16 = (_Float16*)carve((size_t)2 * NT_ * H2_ * sizeof(_Float16));
    float*    em     = (float*)   carve((size_t)NT_ * T_ * sizeof(float));

    pack_weights<<<(2 * G_ * E_) / 256, 256, 0, stream>>>(Wih_f, Wih_b, Whh_f, Whh_b,
                                                          Wih16, Whh16);
    input_proj<<<dim3(NT_ / 16, 8), 256, 0, stream>>>(x, embed, Wih16, b_f, b_b, xg16);
    lstm_kernel<<<16, 256, 0, stream>>>(xg16, Whh16, hbuf16);
    emissions_kernel<<<(NT_ * T_ + 255) / 256, 256, 0, stream>>>(hbuf16, W_out, b_out, em);
    viterbi_kernel<<<B_, 32, 0, stream>>>(em, mask, start, trans, endt, (float*)d_out);
}
